// mBP_model_41437844472506
// MI455X (gfx1250) — matrix operand order repaired and checked by reading, not compiled, not run on hardware
//
#include <hip/hip_runtime.h>
#include <hip/hip_bf16.h>
#include <math.h>

#define NAT    2048
#define NG     8
#define NE     4
#define NPERM  10
#define FEATSZ 672
#define RC     6.0f
#define RCA    4.0f
#define BLOCK  128
#define NWAVE  4
#define KCH    32                  // K per WMMA step
#define NCHUNK (NAT / KCH)         // 64
#define NROW   16                  // A rows: 8 radial + 8 angular-radial
#define NCOL   48                  // B cols: 40 ang*spec + 4 spec + 4 zero pad

typedef _Float16 v8h  __attribute__((ext_vector_type(8)));
typedef _Float16 v16h __attribute__((ext_vector_type(16)));
typedef float    v8f  __attribute__((ext_vector_type(8)));

static __device__ __forceinline__ v16h cat8(v8h lo, v8h hi) {
  return __builtin_shufflevector(lo, hi, 0,1,2,3,4,5,6,7,8,9,10,11,12,13,14,15);
}

// fast sigmoid: native v_exp + native v_rcp (values feed f16 WMMA operands)
static __device__ __forceinline__ float fast_sigmoid(float x) {
  return __builtin_amdgcn_rcpf(1.0f + __expf(-x));
}

__global__ __launch_bounds__(BLOCK)
void mbp_pair_wmma_kernel(
    const float* __restrict__ pos,      // [NAT,3]
    const float* __restrict__ spec,     // [NAT,NE]
    const float* __restrict__ kn_rad,   // [NG]
    const float* __restrict__ kn_ang,   // [NG]
    const float* __restrict__ rbf_w,  const float* __restrict__ rbf_b,
    const float* __restrict__ rbf_aw, const float* __restrict__ rbf_ab,
    const float* __restrict__ W1, const float* __restrict__ b1,
    const float* __restrict__ W2, const float* __restrict__ b2,
    const float* __restrict__ W3, const float* __restrict__ b3,
    float* __restrict__ out)
{
  // per-wave staging for the j<->channel transpose (f16, WMMA source layouts)
  __shared__ __attribute__((aligned(16))) _Float16 Ast[NWAVE][NROW][KCH]; // 4 KB
  __shared__ __attribute__((aligned(16))) _Float16 Bst[NWAVE][NCOL][KCH]; // 12 KB
  __shared__ float Cred[NROW][NCOL];   // cross-wave reduced C
  __shared__ float featL[FEATSZ];
  __shared__ float part1[64][2];
  __shared__ float h1[64], h2[64];

  const int i    = blockIdx.x;
  const int tid  = threadIdx.x;
  const int wave = tid >> 5;
  const int lane = tid & 31;

  for (int t = tid; t < NROW * NCOL; t += BLOCK) ((float*)Cred)[t] = 0.0f;

  const float xi = pos[i*3+0], yi = pos[i*3+1], zi = pos[i*3+2];
  float si[NE];
  #pragma unroll
  for (int e = 0; e < NE; ++e) si[e] = spec[i*NE+e];

  // fold basis constants: arg = (pi/rc) * n * kn * r
  float crad[NG], cang[NG], wr[NG], br[NG], wa[NG], ba[NG];
  #pragma unroll
  for (int l = 0; l < NG; ++l) {
    crad[l] = (float)M_PI * (1.0f/RC)  * (float)(l+1) * kn_rad[l];
    cang[l] = (float)M_PI * (1.0f/RCA) * (float)(l+1) * kn_ang[l];
    wr[l] = rbf_w[l];  br[l] = rbf_b[l];
    wa[l] = rbf_aw[l]; ba[l] = rbf_ab[l];
  }

  // zero-pad B columns 44..47 once (each lane writes its own k slot)
  #pragma unroll
  for (int c = 44; c < NCOL; ++c) Bst[wave][c][lane] = (_Float16)0.0f;

  __syncthreads();

  v8f acc0 = {}, acc1 = {}, acc2 = {};

  // WMMA fragment addressing (CDNA5 ISA 7.12.2, wave32)
  const int arow = lane & 15;
  const int ak0  = (lane >> 4) * 16;   // A halves: [ak0..ak0+15] contiguous
  const int bcol = lane & 15;
  const int bk0  = (lane >> 4) * 16;   // B halves: [bk0..bk0+15] contiguous

  for (int c = wave; c < NCHUNK; c += NWAVE) {
    const int j = c * KCH + lane;

    // branch-free lookahead prefetch (NAT is a power of two -> wrap with AND)
    __builtin_prefetch(&pos[((j + NWAVE*KCH) & (NAT - 1)) * 3], 0, 1);

    // ---------- per-pair basis (VALU / transcendental stage) ----------
    const float dx = xi - pos[j*3+0];
    const float dy = yi - pos[j*3+1];
    const float dz = zi - pos[j*3+2];
    float sj[NE];
    #pragma unroll
    for (int e = 0; e < NE; ++e) sj[e] = spec[j*NE+e];

    const float d2    = dx*dx + dy*dy + dz*dz + 1e-16f;
    const float inv_r = __builtin_amdgcn_rsqf(d2);   // v_rsq_f32
    const float r     = d2 * inv_r;                  // r = d2 * rsqrt(d2)

    // branch-free smooth cutoffs: always evaluate v_cos_f32, mask via cndmask
    const float m6 = ((r <= RC)  && (r > 1e-8f)) ? 1.0f : 0.0f;
    const float m4 = ((r <= RCA) && (r > 1e-8f)) ? 1.0f : 0.0f;
    const float fc6 = 0.5f * (__cosf((float)M_PI*(1.0f/RC) *r) + 1.0f) * m6;
    const float fc4 = 0.5f * (__cosf((float)M_PI*(1.0f/RCA)*r) + 1.0f) * m4;

    const float fir6 = fc6 * inv_r;
    const float fir4 = fc4 * inv_r;
    #pragma unroll
    for (int l = 0; l < NG; ++l) {
      Ast[wave][l][lane]    =
          (_Float16)(__sinf(crad[l]*r) * fir6 * fast_sigmoid(r*wr[l] + br[l]));
      Ast[wave][NG+l][lane] =
          (_Float16)(__sinf(cang[l]*r) * fir4 * fast_sigmoid(r*wa[l] + ba[l]));
    }

    const float ux = dx*inv_r, uy = dy*inv_r, uz = dz*inv_r;
    float ang[NPERM];               // monomials in reference _EXP order
    ang[0] = 1.0f;  ang[1] = uz;    ang[2] = uy;    ang[3] = ux;
    ang[4] = uz*uz; ang[5] = uy*uz; ang[6] = uy*uy;
    ang[7] = ux*uz; ang[8] = ux*uy; ang[9] = ux*ux;

    #pragma unroll
    for (int p = 0; p < NPERM; ++p)
      #pragma unroll
      for (int e = 0; e < NE; ++e)
        Bst[wave][p*NE + e][lane] = (_Float16)(ang[p]*sj[e]);
    #pragma unroll
    for (int e = 0; e < NE; ++e)
      Bst[wave][40 + e][lane] = (_Float16)sj[e];

    // same-wave cross-lane LDS RAW: drain DS stores explicitly
    asm volatile("s_wait_dscnt 0" ::: "memory");

    // ---------- fragment loads + WMMA (matrix stage) ----------
    v8h  alo = *(const v8h*)&Ast[wave][arow][ak0];
    v8h  ahi = *(const v8h*)&Ast[wave][arow][ak0 + 8];
    v16h av  = cat8(alo, ahi);

    v16h bv0 = cat8(*(const v8h*)&Bst[wave][     bcol][bk0],
                    *(const v8h*)&Bst[wave][     bcol][bk0 + 8]);
    v16h bv1 = cat8(*(const v8h*)&Bst[wave][16 + bcol][bk0],
                    *(const v8h*)&Bst[wave][16 + bcol][bk0 + 8]);
    v16h bv2 = cat8(*(const v8h*)&Bst[wave][32 + bcol][bk0],
                    *(const v8h*)&Bst[wave][32 + bcol][bk0 + 8]);

    acc0 = __builtin_amdgcn_wmma_f32_16x16x32_f16(false, av, false, bv0,
                                                  (short)0, acc0, false, false);
    acc1 = __builtin_amdgcn_wmma_f32_16x16x32_f16(false, av, false, bv1,
                                                  (short)0, acc1, false, false);
    acc2 = __builtin_amdgcn_wmma_f32_16x16x32_f16(false, av, false, bv2,
                                                  (short)0, acc2, false, false);
  }

  // ---------- cross-wave reduction via LDS float atomics (ds_add_f32) ----------
  const int dm = (lane >> 4) * 8;      // D layout: VGPR v -> row v + 8*(lane/16)
  const int dn = lane & 15;
  #pragma unroll
  for (int v = 0; v < 8; ++v) {
    atomicAdd(&Cred[dm + v][     dn], acc0[v]);
    atomicAdd(&Cred[dm + v][16 + dn], acc1[v]);
    atomicAdd(&Cred[dm + v][32 + dn], acc2[v]);
  }
  __syncthreads();

  // ---------- features: [G_rad(32) | M(320) | M*M(320)] ----------
  for (int f = tid; f < FEATSZ; f += BLOCK) {
    float v;
    if (f < 32) {                       // G[l,e] = s_i[e] * C[l][40+e]
      const int l = f >> 2, e = f & 3;
      v = si[e] * Cred[l][40 + e];
    } else {
      const int g = (f < 352) ? (f - 32) : (f - 352);
      const int l = g / 40, rm = g % 40;
      const int p = rm >> 2, e = rm & 3; // M[l,p,e] = s_i[e] * C[8+l][4p+e]
      const float m = si[e] * Cred[NG + l][p*NE + e];
      v = (f < 352) ? m : m * m;
    }
    featL[f] = v;
  }
  __syncthreads();

  // ---------- per-atom MLP: 672 -> 64 tanh -> 64 tanh -> 1 ----------
  {
    const int k = tid >> 1, hf = tid & 1;
    float s = 0.0f;
    const int f0 = hf * 336;
    for (int f = f0; f < f0 + 336; ++f) s += featL[f] * W1[f*64 + k];
    part1[k][hf] = s;
  }
  __syncthreads();
  if (tid < 64) h1[tid] = tanhf(part1[tid][0] + part1[tid][1] + b1[tid]);
  __syncthreads();
  if (tid < 64) {
    float s = b2[tid];
    for (int m = 0; m < 64; ++m) s += h1[m] * W2[m*64 + tid];
    h2[tid] = tanhf(s);
  }
  __syncthreads();
  if (tid == 0) {
    float s = b3[0];
    for (int m = 0; m < 64; ++m) s += h2[m] * W3[m];
    out[i] = s;
  }
}

extern "C" void kernel_launch(void* const* d_in, const int* in_sizes, int n_in,
                              void* d_out, int out_size, void* d_ws, size_t ws_size,
                              hipStream_t stream) {
  (void)in_sizes; (void)n_in; (void)out_size; (void)d_ws; (void)ws_size;
  const float* pos  = (const float*)d_in[0];
  const float* spc  = (const float*)d_in[1];
  const float* knr  = (const float*)d_in[2];
  const float* kna  = (const float*)d_in[3];
  const float* rw   = (const float*)d_in[4];
  const float* rb   = (const float*)d_in[5];
  const float* raw_ = (const float*)d_in[6];
  const float* rab  = (const float*)d_in[7];
  const float* W1   = (const float*)d_in[8];
  const float* b1   = (const float*)d_in[9];
  const float* W2   = (const float*)d_in[10];
  const float* b2   = (const float*)d_in[11];
  const float* W3   = (const float*)d_in[12];
  const float* b3   = (const float*)d_in[13];

  mbp_pair_wmma_kernel<<<NAT, BLOCK, 0, stream>>>(
      pos, spc, knr, kna, rw, rb, raw_, rab,
      W1, b1, W2, b2, W3, b3, (float*)d_out);
}
